// ConditionalAffineCoupling_69483980914691
// MI455X (gfx1250) — matrix operand order repaired and checked by reading, not compiled, not run on hardware
//
#include <hip/hip_runtime.h>
#include <hip/hip_bf16.h>

typedef __attribute__((ext_vector_type(16))) _Float16 v16h;
typedef __attribute__((ext_vector_type(8)))  _Float16 v8h;
typedef __attribute__((ext_vector_type(8)))  float    v8f;

#define WAVES   8
#define THREADS (WAVES * 32)

// Weight fragment storage (f16, WMMA B-operand lane order)
//   layer1: W1 padded 24->32 K, 1 kstep x 8 ntiles  = 8  frags * 512 elems = 4096
//   layer2: W2 128x128,        4 ksteps x 8 ntiles  = 32 frags * 512 elems = 16384
//   layer3: W3 128x32,         4 ksteps x 2 ntiles  = 8  frags * 512 elems = 4096
#define L1_OFF  0
#define L2_OFF  4096
#define L3_OFF  20480
#define W_ELEMS 24576

__device__ __forceinline__ v8f wmma_f16(v16h a, v16h b, v8f c) {
    // (neg_a, A, neg_b, B, c_mod, C, reuse_a, reuse_b)
    return __builtin_amdgcn_wmma_f32_16x16x32_f16(false, a, false, b, (short)0, c, false, false);
}

// Load one B fragment (16 f16 per lane) from prepacked LDS: two ds_load_b128.
__device__ __forceinline__ v16h load_bfrag(const _Float16* fragBase, int lane) {
    const v8h* p = (const v8h*)(fragBase + lane * 16);
    v8h lo = p[0];
    v8h hi = p[1];
    v16h r;
#pragma unroll
    for (int i = 0; i < 8; ++i) { r[i] = lo[i]; r[i + 8] = hi[i]; }
    return r;
}

// Load one A fragment (16x32 slice of a row-major 16x128 f16 LDS tile).
// A layout: lane m (group g): elems 0..7 <-> K = ks*32 + g*8 + 0..7
//                              elems 8..15 <-> K = ks*32 + 16 + g*8 + 0..7
__device__ __forceinline__ v16h load_afrag128(const _Float16* h, int m, int g, int ks) {
    const v8h* p0 = (const v8h*)(h + m * 128 + ks * 32 + g * 8);
    const v8h* p1 = (const v8h*)(h + m * 128 + ks * 32 + 16 + g * 8);
    v8h lo = *p0;
    v8h hi = *p1;
    v16h r;
#pragma unroll
    for (int i = 0; i < 8; ++i) { r[i] = lo[i]; r[i + 8] = hi[i]; }
    return r;
}

__global__ __launch_bounds__(THREADS)
void coupling_kernel(const float* __restrict__ x,  const float* __restrict__ ctx,
                     const float* __restrict__ W1, const float* __restrict__ b1,
                     const float* __restrict__ W2, const float* __restrict__ b2,
                     const float* __restrict__ W3, const float* __restrict__ b3,
                     float* __restrict__ yOut, float* __restrict__ logDet, int nRows)
{
    __shared__ alignas(16) _Float16 sW[W_ELEMS];
    __shared__ float sBias[288];                      // b1[0:128], b2[128:256], b3[256:288]
    __shared__ alignas(16) char sScratch[WAVES][4096]; // per-wave h (16x128 f16) / st (16x32 f32)

    const int tid = threadIdx.x;

    // ---- prepack weights f32->f16 into WMMA B-fragment lane order ----
    for (int idx = tid; idx < W_ELEMS; idx += THREADS) {
        const float* src; int NT, Keff, N, rel;
        if (idx < L2_OFF)      { src = W1; NT = 8; Keff = 24;  N = 128; rel = idx; }
        else if (idx < L3_OFF) { src = W2; NT = 8; Keff = 128; N = 128; rel = idx - L2_OFF; }
        else                   { src = W3; NT = 2; Keff = 128; N = 32;  rel = idx - L3_OFF; }
        int fi = rel >> 9;          // fragment index within layer
        int le = rel & 511;
        int l  = le >> 4;           // lane 0..31
        int e  = le & 15;           // element within v16h
        int ks = fi / NT;
        int nt = fi - ks * NT;
        int gg = l >> 4;
        int k  = ks * 32 + gg * 8 + (e & 7) + ((e >> 3) << 4);   // B layout: elem e of lane group g <-> K = g*8+(e&7)+16*(e>>3)
        int n  = nt * 16 + (l & 15);
        float v = (k < Keff) ? src[k * N + n] : 0.0f;
        sW[idx] = (_Float16)v;
    }
    if (tid < 128) sBias[tid]       = b1[tid];
    if (tid < 128) sBias[128 + tid] = b2[tid];
    if (tid < 32)  sBias[256 + tid] = b3[tid];
    __syncthreads();

    const int wave = tid >> 5;
    const int lane = tid & 31;
    const int m    = lane & 15;   // row within 16-row tile / N within tile
    const int g    = lane >> 4;   // half-wave group

    _Float16* hbuf = (_Float16*)sScratch[wave];
    float*    stf  = (float*)sScratch[wave];

    const int  nTiles     = (nRows + 15) >> 4;
    const long wavesTotal = (long)gridDim.x * WAVES;
    const int  iters      = (int)(((long)nTiles + wavesTotal - 1) / wavesTotal); // uniform

    for (int it = 0; it < iters; ++it) {
        long tile = (long)it * wavesTotal + (long)blockIdx.x * WAVES + wave;
        long row  = tile * 16 + m;
        long rowc = (row < nRows) ? row : (long)(nRows - 1);
        bool active = (row < nRows);

        // ---- load x half-row (16 floats): lanes g=0 cols 0..15, g=1 cols 16..31 ----
        const float4* xp = (const float4*)(x + rowc * 32 + g * 16);
        float4 xv0 = xp[0], xv1 = xp[1], xv2 = xp[2], xv3 = xp[3];
        float xc[16] = { xv0.x, xv0.y, xv0.z, xv0.w, xv1.x, xv1.y, xv1.z, xv1.w,
                         xv2.x, xv2.y, xv2.z, xv2.w, xv3.x, xv3.y, xv3.z, xv3.w };

        // ---- build layer-1 A fragment in registers ----
        // g=0: K0..7  = x_masked[0..7]  (even cols 0..14),  K16..23 = context[0..7]
        // g=1: K8..15 = x_masked[8..15] (even cols 16..30), K24..31 = 0 (pad)
        v16h a1;
#pragma unroll
        for (int e = 0; e < 8; ++e) a1[e] = (_Float16)xc[2 * e];
        if (g == 0) {
            const float4* cp = (const float4*)(ctx + rowc * 8);
            float4 c0 = cp[0], c1 = cp[1];
            a1[8]  = (_Float16)c0.x; a1[9]  = (_Float16)c0.y;
            a1[10] = (_Float16)c0.z; a1[11] = (_Float16)c0.w;
            a1[12] = (_Float16)c1.x; a1[13] = (_Float16)c1.y;
            a1[14] = (_Float16)c1.z; a1[15] = (_Float16)c1.w;
        } else {
#pragma unroll
            for (int e = 8; e < 16; ++e) a1[e] = (_Float16)0.0f;
        }

        // ---- layer 1: (16x32) @ (32x128), relu, store h1 f16 row-major ----
#pragma unroll
        for (int nt = 0; nt < 8; ++nt) {
            v8f acc = {0.f, 0.f, 0.f, 0.f, 0.f, 0.f, 0.f, 0.f};
            v16h bf = load_bfrag(sW + L1_OFF + nt * 512, lane);
            acc = wmma_f16(a1, bf, acc);
            int col = nt * 16 + m;
            float bias = sBias[col];
#pragma unroll
            for (int r = 0; r < 8; ++r)
                hbuf[(r + g * 8) * 128 + col] = (_Float16)fmaxf(acc[r] + bias, 0.0f);
        }
        __syncthreads();

        // ---- layer 2: (16x128) @ (128x128) ----
        v16h a2[4];
#pragma unroll
        for (int ks = 0; ks < 4; ++ks) a2[ks] = load_afrag128(hbuf, m, g, ks);
        __syncthreads();

#pragma unroll
        for (int nt = 0; nt < 8; ++nt) {
            v8f acc = {0.f, 0.f, 0.f, 0.f, 0.f, 0.f, 0.f, 0.f};
#pragma unroll
            for (int ks = 0; ks < 4; ++ks) {
                v16h bf = load_bfrag(sW + L2_OFF + (ks * 8 + nt) * 512, lane);
                acc = wmma_f16(a2[ks], bf, acc);
            }
            int col = nt * 16 + m;
            float bias = sBias[128 + col];
#pragma unroll
            for (int r = 0; r < 8; ++r)
                hbuf[(r + g * 8) * 128 + col] = (_Float16)fmaxf(acc[r] + bias, 0.0f);
        }
        __syncthreads();

        // ---- layer 3: (16x128) @ (128x32), store st as f32 ----
        v16h a3[4];
#pragma unroll
        for (int ks = 0; ks < 4; ++ks) a3[ks] = load_afrag128(hbuf, m, g, ks);
        __syncthreads();

#pragma unroll
        for (int nt = 0; nt < 2; ++nt) {
            v8f acc = {0.f, 0.f, 0.f, 0.f, 0.f, 0.f, 0.f, 0.f};
#pragma unroll
            for (int ks = 0; ks < 4; ++ks) {
                v16h bf = load_bfrag(sW + L3_OFF + (ks * 2 + nt) * 512, lane);
                acc = wmma_f16(a3[ks], bf, acc);
            }
            int col = nt * 16 + m;
            float bias = sBias[256 + col];
#pragma unroll
            for (int r = 0; r < 8; ++r)
                stf[(r + g * 8) * 32 + col] = acc[r] + bias;
        }
        __syncthreads();

        // ---- epilogue: s=tanh, y_odd = x_odd*exp(s)+t, log_det = sum(s) ----
        const float* srow = stf + m * 32;
        float part = 0.0f;
        float yv[16];
#pragma unroll
        for (int j = 0; j < 8; ++j) {
            float s = tanhf(srow[g * 8 + j]);
            float t = srow[16 + g * 8 + j];
            part += s;
            yv[2 * j]     = xc[2 * j];
            yv[2 * j + 1] = xc[2 * j + 1] * __expf(s) + t;
        }
        float tot = part + __shfl_xor(part, 16, 32);
        if (active) {
            float4* yp = (float4*)(yOut + row * 32 + g * 16);
            yp[0] = make_float4(yv[0],  yv[1],  yv[2],  yv[3]);
            yp[1] = make_float4(yv[4],  yv[5],  yv[6],  yv[7]);
            yp[2] = make_float4(yv[8],  yv[9],  yv[10], yv[11]);
            yp[3] = make_float4(yv[12], yv[13], yv[14], yv[15]);
            if (g == 0) logDet[row] = tot;
        }
        __syncthreads();  // protect scratch reuse across tile iterations (uniform)
    }
}

extern "C" void kernel_launch(void* const* d_in, const int* in_sizes, int n_in,
                              void* d_out, int out_size, void* d_ws, size_t ws_size,
                              hipStream_t stream) {
    const float* x   = (const float*)d_in[0];
    const float* ctx = (const float*)d_in[1];
    const float* W1  = (const float*)d_in[2];
    const float* b1  = (const float*)d_in[3];
    const float* W2  = (const float*)d_in[4];
    const float* b2  = (const float*)d_in[5];
    const float* W3  = (const float*)d_in[6];
    const float* b3  = (const float*)d_in[7];

    const int nRows = in_sizes[0] / 32;           // B
    float* y  = (float*)d_out;                    // (B,32) then log_det (B,)
    float* ld = y + (size_t)nRows * 32;

    int nTiles = (nRows + 15) / 16;
    int blocks = (nTiles + WAVES - 1) / WAVES;
    if (blocks > 2048) blocks = 2048;             // grid-stride loop amortizes weight prepack
    if (blocks < 1) blocks = 1;

    coupling_kernel<<<blocks, THREADS, 0, stream>>>(x, ctx, W1, b1, W2, b2, W3, b3, y, ld, nRows);
}
